// CheMoE_PDG_9663676416496
// MI455X (gfx1250) — matrix operand order repaired and checked by reading, not compiled, not run on hardware
//
#include <hip/hip_runtime.h>
#include <hip/hip_bf16.h>

typedef __attribute__((ext_vector_type(16))) _Float16 v16h;
typedef __attribute__((ext_vector_type(8)))  float    v8f;

#define NGENES 10716
#define EMB    128
#define NB     8
#define NEXP   4

// ---------------------------------------------------------------- utilities

__global__ void zero_kernel(float* p, int n) {
    int i = blockIdx.x * blockDim.x + threadIdx.x;
    if (i < n) p[i] = 0.0f;
}

__device__ __forceinline__ float block_reduce_1024(float v, float* scratch, int t) {
    scratch[t] = v; __syncthreads();
    for (int s = 512; s > 0; s >>= 1) { if (t < s) scratch[t] += scratch[t + s]; __syncthreads(); }
    float r = scratch[0]; __syncthreads();
    return r;
}

__device__ __forceinline__ float block_reduce_256(float v, float* scratch, int t) {
    scratch[t] = v; __syncthreads();
    for (int s = 128; s > 0; s >>= 1) { if (t < s) scratch[t] += scratch[t + s]; __syncthreads(); }
    float r = scratch[0]; __syncthreads();
    return r;
}

// -------------------------------------------- encoder: Linear->LN->ReLU->Linear->LN
// grid = 8 (one block per batch row), block = 1024 threads. d_h=256, d_out=128.
__global__ __launch_bounds__(1024) void encode_kernel(
    const float* __restrict__ x, int d_in,
    const float* __restrict__ W1, const float* __restrict__ b1,
    const float* __restrict__ g1, const float* __restrict__ be1,
    const float* __restrict__ W2, const float* __restrict__ b2,
    const float* __restrict__ g2, const float* __restrict__ be2,
    float* __restrict__ out)
{
    __shared__ float red[1024];
    __shared__ float sh[256];
    int b = blockIdx.x, t = threadIdx.x;
    int j = t & 255, p = t >> 8;                 // 4 K-partials per output
    const float* xb = x + (size_t)b * d_in;

    float acc = 0.0f;
    for (int i = p; i < d_in; i += 4) acc = fmaf(xb[i], W1[(size_t)i * 256 + j], acc);
    red[t] = acc; __syncthreads();

    float hval = 0.0f;
    if (t < 256) {
        hval = red[j] + red[256 + j] + red[512 + j] + red[768 + j] + b1[j];
        sh[j] = hval;
    }
    __syncthreads();

    float v   = (t < 256) ? sh[t] : 0.0f;
    float sum = block_reduce_1024(v, red, t);
    float ssq = block_reduce_1024(v * v, red, t);
    float mean = sum * (1.0f / 256.0f);
    float rstd = rsqrtf(ssq * (1.0f / 256.0f) - mean * mean + 1e-5f);
    if (t < 256) sh[j] = fmaxf((hval - mean) * rstd * g1[j] + be1[j], 0.0f);
    __syncthreads();

    // layer 2: 8 K-partials per output
    int j2 = t & 127, p2 = t >> 7;
    float a2 = 0.0f;
    for (int k = p2; k < 256; k += 8) a2 = fmaf(sh[k], W2[k * 128 + j2], a2);
    __syncthreads();
    red[t] = a2; __syncthreads();
    float o = 0.0f;
    if (t < 128) {
        o = b2[t];
        #pragma unroll
        for (int q = 0; q < 8; q++) o += red[q * 128 + t];
    }
    __syncthreads();
    float v2   = (t < 128) ? o : 0.0f;
    float sum2 = block_reduce_1024(v2, red, t);
    float ssq2 = block_reduce_1024(v2 * v2, red, t);
    float mean2 = sum2 * (1.0f / 128.0f);
    float rstd2 = rsqrtf(ssq2 * (1.0f / 128.0f) - mean2 * mean2 + 1e-5f);
    if (t < 128) out[b * 128 + t] = (o - mean2) * rstd2 * g2[t] + be2[t];
}

// -------------------------------------------- gate: gf assembly + MLP + top-2 softmax
__global__ __launch_bounds__(256) void gate_kernel(
    const float* __restrict__ drugW, const float* __restrict__ basalW,
    const float* __restrict__ cellEmb, const int* __restrict__ cellIdx,
    const float* __restrict__ Wg1, const float* __restrict__ bg1,
    const float* __restrict__ gg1, const float* __restrict__ beg1,
    const float* __restrict__ Wg2, const float* __restrict__ bg2,
    float* __restrict__ gfW, float* __restrict__ weightsW)
{
    __shared__ float sgf[NB][384];
    __shared__ float sgh[128];
    __shared__ float slog[4];
    __shared__ float scratch[256];
    int t = threadIdx.x;

    for (int idx = t; idx < NB * 384; idx += 256) {
        int b = idx / 384, k = idx % 384;
        float v;
        if (k < 128)      v = drugW[b * 128 + k];
        else if (k < 256) v = cellEmb[cellIdx[b] * 128 + (k - 128)];
        else              v = basalW[b * 128 + (k - 256)];
        sgf[b][k] = v;
        gfW[b * 384 + k] = v;
    }
    __syncthreads();

    for (int b = 0; b < NB; b++) {
        float hv = 0.0f;
        if (t < 128) {
            hv = bg1[t];
            for (int k = 0; k < 384; k++) hv = fmaf(sgf[b][k], Wg1[k * 128 + t], hv);
        }
        float v   = (t < 128) ? hv : 0.0f;
        float sum = block_reduce_256(v, scratch, t);
        float ssq = block_reduce_256(v * v, scratch, t);
        float mean = sum * (1.0f / 128.0f);
        float rstd = rsqrtf(ssq * (1.0f / 128.0f) - mean * mean + 1e-5f);
        if (t < 128) sgh[t] = fmaxf((hv - mean) * rstd * gg1[t] + beg1[t], 0.0f);
        __syncthreads();
        if (t < 4) {
            float lg = bg2[t];
            for (int k = 0; k < 128; k++) lg = fmaf(sgh[k], Wg2[k * 4 + t], lg);
            slog[t] = lg;
        }
        __syncthreads();
        if (t == 0) {
            int i0 = 0;
            for (int q = 1; q < 4; q++) if (slog[q] > slog[i0]) i0 = q;
            int i1 = -1;
            for (int q = 0; q < 4; q++) if (q != i0 && (i1 < 0 || slog[q] > slog[i1])) i1 = q;
            float e1 = __expf(slog[i1] - slog[i0]);
            float w0 = 1.0f / (1.0f + e1);
            for (int q = 0; q < 4; q++) weightsW[b * 4 + q] = 0.0f;
            weightsW[b * 4 + i0] = w0;
            weightsW[b * 4 + i1] = e1 * w0;
        }
        __syncthreads();
    }
}

// -------------------------------------------- gfterm[e,b,:] = gf[b] @ W1[e][:384] + b1[e]
__global__ __launch_bounds__(256) void gfterm_kernel(
    const float* __restrict__ gfW, const float* __restrict__ eW1,
    const float* __restrict__ eB1, float* __restrict__ gftermW)
{
    int e = blockIdx.x >> 3, b = blockIdx.x & 7, n = threadIdx.x;
    float acc = eB1[e * 256 + n];
    const float* gfb = gfW + b * 384;
    const float* w   = eW1 + (size_t)e * 512 * 256 + n;
    for (int k = 0; k < 384; k++) acc = fmaf(gfb[k], w[(size_t)k * 256], acc);
    gftermW[((e * 8) + b) * 256 + n] = acc;
}

// -------------------------------------------- WMMA fragment loaders (LDS, f16)
// A layout (16x32 f16): lane l -> row m=l&15; elems 0..7 = K 8h..8h+7, 8..15 = K 16+8h.. (h=l>>4)
__device__ __forceinline__ v16h load_af(const _Float16* base, int rowStride, int m, int hh, int kbase) {
    const _Float16* p = base + m * rowStride + kbase + 8 * hh;
    union { v16h v; uint4 q[2]; } u;
    u.q[0] = *reinterpret_cast<const uint4*>(p);
    u.q[1] = *reinterpret_cast<const uint4*>(p + 16);
    return u.v;
}
// B layout (32x16 f16): lane l -> col n=l&15; elems 0..7 = K 8h..8h+7, 8..15 = K 16+8h..
__device__ __forceinline__ v16h load_bf(const _Float16* base, int rowStride, int n, int hh, int kbase) {
    const _Float16* p = base + n * rowStride + kbase + 8 * hh;
    union { v16h v; uint4 q[2]; } u;
    u.q[0] = *reinterpret_cast<const uint4*>(p);
    u.q[1] = *reinterpret_cast<const uint4*>(p + 16);
    return u.v;
}

// -------------------------------------------- main fused expert kernel
// grid = (670 gene tiles, 4 experts), block = 256 threads = 8 waves (wave w -> batch b=w).
__global__ __launch_bounds__(256, 1) void moe_main_kernel(
    const float* __restrict__ geneEmb,
    const float* __restrict__ eW1, const float* __restrict__ eW2,
    const float* __restrict__ eB2,
    const float* __restrict__ eG1, const float* __restrict__ eBe1,
    const float* __restrict__ eG2, const float* __restrict__ eBe2,
    const float* __restrict__ eW3, const float* __restrict__ eB3,
    const float* __restrict__ gftermW, const float* __restrict__ weightsW,
    float* __restrict__ out)
{
    constexpr int W1T_RS = 136;  // halves/row for [256][128]+pad
    constexpr int W2T_RS = 264;  // halves/row for [128][256]+pad
    constexpr int GT_RS  = 260;  // floats/row for [16][256]+pad

    __shared__ __align__(16) _Float16 sGene[16 * 136];
    __shared__ __align__(16) _Float16 sWt[256 * 136];   // phase A: W1x^T; phase B: W2^T (128*264 fits)
    __shared__ __align__(16) float sGT[16 * GT_RS];     // geneterm, fp32
    __shared__ float sGF[NB * 256];                     // gfterm (incl. b1), fp32
    __shared__ float g1s[256], be1s[256];
    __shared__ float g2s[128], be2s[128], w3s[128], b2s[128];

    const int e  = blockIdx.y;
    const int g0 = blockIdx.x * 16;
    const int t  = threadIdx.x;
    const int lane = t & 31, wv = t >> 5;
    const int l15 = lane & 15, hh = lane >> 4;

    if (t == 0 && g0 + 16 < NGENES)
        __builtin_prefetch(geneEmb + (size_t)(g0 + 16) * EMB, 0, 3);

    // ---- stage A loads
    for (int idx = t; idx < 16 * 128; idx += 256) {
        int m = idx >> 7, k = idx & 127;
        int g = g0 + m;
        float v = (g < NGENES) ? geneEmb[(size_t)g * EMB + k] : 0.0f;
        sGene[m * W1T_RS + k] = (_Float16)v;
    }
    for (int idx = t; idx < 256 * 128; idx += 256) {
        int n = idx >> 7, k = idx & 127;   // W1x^T[n][k] = W1[e][384+k][n]
        sWt[n * W1T_RS + k] = (_Float16)eW1[((size_t)e * 512 + 384 + k) * 256 + n];
    }
    for (int idx = t; idx < NB * 256; idx += 256) sGF[idx] = gftermW[(size_t)(e * 8) * 256 + idx];
    g1s[t]  = eG1[e * 256 + t];
    be1s[t] = eBe1[e * 256 + t];
    if (t < 128) {
        g2s[t]  = eG2[e * 128 + t];
        be2s[t] = eBe2[e * 128 + t];
        w3s[t]  = eW3[e * 128 + t];
        b2s[t]  = eB2[e * 128 + t];
    }
    __syncthreads();

    // ---- layer 1: geneterm[16x256] = geneEmb_tile(f16) x W1x(f16), shared across batch
    {
        v16h afr[4];
        #pragma unroll
        for (int s = 0; s < 4; s++) afr[s] = load_af(sGene, W1T_RS, l15, hh, 32 * s);
        #pragma unroll
        for (int ni = 0; ni < 2; ni++) {
            int nt = wv * 2 + ni;            // 8 waves x 2 = 16 N-tiles of 16
            v8f acc = {};
            #pragma unroll
            for (int s = 0; s < 4; s++) {
                v16h bfr = load_bf(sWt, W1T_RS, nt * 16 + l15, hh, 32 * s);
                acc = __builtin_amdgcn_wmma_f32_16x16x32_f16(false, afr[s], false, bfr,
                                                             (short)0, acc, false, false);
            }
            #pragma unroll
            for (int v = 0; v < 8; v++)
                sGT[(v + 8 * hh) * GT_RS + nt * 16 + l15] = acc[v];
        }
    }
    __syncthreads();

    // ---- reload shared weight buffer with W2^T
    for (int idx = t; idx < 256 * 128; idx += 256) {
        int n = idx & 127, k = idx >> 7;     // W2^T[n][k] = W2[e][k][n]
        sWt[n * W2T_RS + k] = (_Float16)eW2[((size_t)e * 256 + k) * 128 + n];
    }
    __syncthreads();

    // ---- per-batch (one wave per b), top-2 sparsity skip (wave-uniform, EXEC stays full)
    const int b = wv;
    const float wt = weightsW[b * 4 + e];
    if (wt != 0.0f) {
        // LN1 stats over 256 per gene row (lane pair l, l^16 covers one row)
        float s1 = 0.0f, s2 = 0.0f;
        const float* rGT = sGT + l15 * GT_RS + hh * 128;
        const float* rGF = sGF + b * 256 + hh * 128;
        for (int i = 0; i < 128; i++) {
            float x = rGT[i] + rGF[i];
            s1 += x; s2 = fmaf(x, x, s2);
        }
        s1 += __shfl_xor(s1, 16, 32);
        s2 += __shfl_xor(s2, 16, 32);
        float mean = s1 * (1.0f / 256.0f);
        float rstd = rsqrtf(s2 * (1.0f / 256.0f) - mean * mean + 1e-5f);

        // build h1 A-fragments: relu(LN(geneterm + gfterm)) -> f16
        v16h h1f[8];
        #pragma unroll
        for (int s = 0; s < 8; s++) {
            v16h f;
            #pragma unroll
            for (int i = 0; i < 16; i++) {
                int K = 32 * s + ((i < 8) ? (8 * hh + i) : (16 + 8 * hh + (i - 8)));
                float x = sGT[l15 * GT_RS + K] + sGF[b * 256 + K];
                float y = fmaxf((x - mean) * rstd * g1s[K] + be1s[K], 0.0f);
                f[i] = (_Float16)y;
            }
            h1f[s] = f;
        }

        // layer 2: [16x256] x [256x128] via 8x8 WMMAs
        float h2r[8][8];
        #pragma unroll
        for (int nt = 0; nt < 8; nt++) {
            v8f acc = {};
            int n2 = nt * 16 + l15;
            #pragma unroll
            for (int s = 0; s < 8; s++) {
                v16h bfr = load_bf(sWt, W2T_RS, n2, hh, 32 * s);
                acc = __builtin_amdgcn_wmma_f32_16x16x32_f16(false, h1f[s], false, bfr,
                                                             (short)0, acc, false, false);
            }
            float bb = b2s[n2];
            #pragma unroll
            for (int v = 0; v < 8; v++) h2r[nt][v] = acc[v] + bb;
        }

        // LN2 + ReLU + W3 dot, per gene row m = v + 8*hh (row spread over 16-lane group)
        const float b3e = eB3[e];
        #pragma unroll
        for (int v = 0; v < 8; v++) {
            float a = 0.0f, q = 0.0f;
            #pragma unroll
            for (int nt = 0; nt < 8; nt++) { float x = h2r[nt][v]; a += x; q = fmaf(x, x, q); }
            a += __shfl_xor(a, 1, 32);  q += __shfl_xor(q, 1, 32);
            a += __shfl_xor(a, 2, 32);  q += __shfl_xor(q, 2, 32);
            a += __shfl_xor(a, 4, 32);  q += __shfl_xor(q, 4, 32);
            a += __shfl_xor(a, 8, 32);  q += __shfl_xor(q, 8, 32);
            float mean2 = a * (1.0f / 128.0f);
            float rstd2 = rsqrtf(q * (1.0f / 128.0f) - mean2 * mean2 + 1e-5f);
            float p = 0.0f;
            #pragma unroll
            for (int nt = 0; nt < 8; nt++) {
                int n2 = nt * 16 + l15;
                float xn = fmaxf((h2r[nt][v] - mean2) * rstd2 * g2s[n2] + be2s[n2], 0.0f);
                p = fmaf(xn, w3s[n2], p);
            }
            p += __shfl_xor(p, 1, 32);
            p += __shfl_xor(p, 2, 32);
            p += __shfl_xor(p, 4, 32);
            p += __shfl_xor(p, 8, 32);
            if (l15 == v) {
                int g = g0 + v + 8 * hh;
                if (g < NGENES)
                    atomicAdd(out + (size_t)b * NGENES + g, wt * (p + b3e));
            }
        }
    }
}

// ---------------------------------------------------------------- launch
extern "C" void kernel_launch(void* const* d_in, const int* in_sizes, int n_in,
                              void* d_out, int out_size, void* d_ws, size_t ws_size,
                              hipStream_t stream) {
    const float* basal_expr = (const float*)d_in[0];
    const float* mol_embed  = (const float*)d_in[1];
    const int*   cell_idx   = (const int*)d_in[2];
    // params.mol
    const float* mW1 = (const float*)d_in[3];  const float* mB1 = (const float*)d_in[4];
    const float* mG1 = (const float*)d_in[5];  const float* mBe1 = (const float*)d_in[6];
    const float* mW2 = (const float*)d_in[7];  const float* mB2 = (const float*)d_in[8];
    const float* mG2 = (const float*)d_in[9];  const float* mBe2 = (const float*)d_in[10];
    // params.basal
    const float* bW1 = (const float*)d_in[11]; const float* bB1 = (const float*)d_in[12];
    const float* bG1 = (const float*)d_in[13]; const float* bBe1 = (const float*)d_in[14];
    const float* bW2 = (const float*)d_in[15]; const float* bB2 = (const float*)d_in[16];
    const float* bG2 = (const float*)d_in[17]; const float* bBe2 = (const float*)d_in[18];
    const float* cellEmb = (const float*)d_in[19];
    const float* geneEmb = (const float*)d_in[20];
    // params.gate
    const float* gW1 = (const float*)d_in[21]; const float* gB1 = (const float*)d_in[22];
    const float* gG1 = (const float*)d_in[23]; const float* gBe1 = (const float*)d_in[24];
    const float* gW2 = (const float*)d_in[25]; const float* gB2 = (const float*)d_in[26];
    // params.experts
    const float* eW1 = (const float*)d_in[27]; const float* eB1 = (const float*)d_in[28];
    const float* eG1 = (const float*)d_in[29]; const float* eBe1 = (const float*)d_in[30];
    const float* eW2 = (const float*)d_in[31]; const float* eB2 = (const float*)d_in[32];
    const float* eG2 = (const float*)d_in[33]; const float* eBe2 = (const float*)d_in[34];
    const float* eW3 = (const float*)d_in[35]; const float* eB3 = (const float*)d_in[36];

    float* out = (float*)d_out;
    float* ws  = (float*)d_ws;
    float* drugW    = ws;                 // 8*128
    float* basalW   = ws + 1024;          // 8*128
    float* gfW      = ws + 2048;          // 8*384
    float* weightsW = ws + 5120;          // 8*4
    float* gftermW  = ws + 5152;          // 4*8*256

    const int outN = NB * NGENES;
    zero_kernel<<<(outN + 255) / 256, 256, 0, stream>>>(out, outN);
    encode_kernel<<<NB, 1024, 0, stream>>>(mol_embed, 2304,
        mW1, mB1, mG1, mBe1, mW2, mB2, mG2, mBe2, drugW);
    encode_kernel<<<NB, 1024, 0, stream>>>(basal_expr, NGENES,
        bW1, bB1, bG1, bBe1, bW2, bB2, bG2, bBe2, basalW);
    gate_kernel<<<1, 256, 0, stream>>>(drugW, basalW, cellEmb, cell_idx,
        gW1, gB1, gG1, gBe1, gW2, gB2, gfW, weightsW);
    gfterm_kernel<<<NEXP * NB, 256, 0, stream>>>(gfW, eW1, eB1, gftermW);

    dim3 grid((NGENES + 15) / 16, NEXP);
    moe_main_kernel<<<grid, 256, 0, stream>>>(geneEmb, eW1, eW2, eB2,
        eG1, eBe1, eG2, eBe2, eW3, eB3, gftermW, weightsW, out);
}
